// InstantNerf_48576080118249
// MI455X (gfx1250) — matrix operand order repaired and checked by reading, not compiled, not run on hardware
//
#include <hip/hip_runtime.h>
#include <hip/hip_bf16.h>

typedef __attribute__((ext_vector_type(16))) _Float16 v16h;
typedef __attribute__((ext_vector_type(8)))  _Float16 v8h;
typedef __attribute__((ext_vector_type(8)))  float    v8f;

#define HT_SIZE 524288

__device__ __forceinline__ v8f wmma_f16(v16h a, v16h b, v8f c) {
  // D = A(16x32 f16) * B(32x16 f16) + C(16x16 f32)
  return __builtin_amdgcn_wmma_f32_16x16x32_f16(false, a, false, b, (short)0, c,
                                                false, false);
}

// ---------------------------------------------------------------------------
// Pre-pass: pack all MLP weights (f32 row-major [fi,fo]) into f16 B-fragments.
// B layout (16-bit, 32x16 tile): lane = n_local + 16*((k_local>>3)&1),
//                                element = (k_local & 7) + 8*(k_local>>4).
// Tile table (each tile = 512 halves = 32 lanes x 16 elements):
//   0-3  : w_d1 (32x64),  nc = t
//   4-5  : w_d2 (64x16),  kc = t-4
//   6-9  : w_c1 (32x64),  nc = t-6
//   10-17: w_c2 (64x64),  nc = (t-10)>>1, kc = (t-10)&1
//   18-19: w_c3 (64x3 padded to 16 cols), kc = t-18
// ---------------------------------------------------------------------------
__global__ void nerf_pack_weights(const float* __restrict__ wd1,
                                  const float* __restrict__ wd2,
                                  const float* __restrict__ wc1,
                                  const float* __restrict__ wc2,
                                  const float* __restrict__ wc3,
                                  _Float16* __restrict__ out) {
  int tid = blockIdx.x * blockDim.x + threadIdx.x;
  if (tid >= 20 * 512) return;
  int t    = tid >> 9;
  int u    = tid & 511;
  int lane = u >> 4;
  int e    = u & 15;
  int kl   = (e & 7) + ((lane >> 4) << 3) + ((e >> 3) << 4);
  int nl   = lane & 15;
  const float* W; int fi, fo, kc = 0, nc = 0;
  if (t < 4)       { W = wd1; fi = 32; fo = 64; nc = t; }
  else if (t < 6)  { W = wd2; fi = 64; fo = 16; kc = t - 4; }
  else if (t < 10) { W = wc1; fi = 32; fo = 64; nc = t - 6; }
  else if (t < 18) { W = wc2; fi = 64; fo = 64; nc = (t - 10) >> 1; kc = (t - 10) & 1; }
  else             { W = wc3; fi = 64; fo = 3;  kc = t - 18; }
  int k = kl + 32 * kc;
  int n = nl + 16 * nc;
  float v = (k < fi && n < fo) ? W[k * fo + n] : 0.0f;
  out[tid] = (_Float16)v;
}

// ---------------------------------------------------------------------------
// Fused kernel: 1 wave = 16 points. Hash-encode -> d1 -> d2(density out) ->
// [density|SH] -> c1 -> c2 -> c3 -> sigmoid(color out).
// ---------------------------------------------------------------------------
__global__ __launch_bounds__(256) void nerf_fused(
    const float* __restrict__ pos, const float* __restrict__ dirp,
    const float* __restrict__ table, const _Float16* __restrict__ wts,
    const float* __restrict__ b_d1, const float* __restrict__ b_d2,
    const float* __restrict__ b_c1, const float* __restrict__ b_c2,
    const float* __restrict__ b_c3,
    float* __restrict__ out_density, float* __restrict__ out_color, int npts) {
  __shared__ __align__(16) _Float16 lds_h[8][1024];  // 2KB per wave
  const int lane = threadIdx.x & 31;
  const int wave = threadIdx.x >> 5;
  const int m    = lane & 15;   // point-row within tile (A), column n (C/D)
  const int h    = lane >> 4;   // half-wave select
  const int base = (blockIdx.x * 8 + wave) * 16;
  if (base >= npts) return;     // wave-uniform
  const int gpt = base + m;
  _Float16* Hl = lds_h[wave];
  const v16h* BW = (const v16h*)wts;

  // ---- hash encoding, built directly as the d1 A-fragment -----------------
  // A(16x32 f16): lane = m + 16*bit3(k), element = (k&7) + 8*(k>=16).
  // With k = 2*lvl+f, lane(h=0) owns levels {0-3,8-11}, lane(h=1) {4-7,12-15},
  // and level j maps to elements {2j, 2j+1}. No cross-lane traffic.
  const float SCALE[16] = {16.f, 21.f, 27.f, 36.f, 48.f, 64.f, 84.f, 111.f,
                           147.f, 194.f, 256.f, 337.f, 445.f, 588.f, 776.f, 1024.f};
  const int SELX[8] = {1,1,1,0,1,0,0,0};
  const int SELY[8] = {1,1,0,1,0,1,0,0};
  const int SELZ[8] = {1,0,1,1,0,0,1,0};
  float px = pos[gpt * 3 + 0], py = pos[gpt * 3 + 1], pz = pos[gpt * 3 + 2];
  v16h aenc;
#pragma unroll
  for (int j = 0; j < 8; ++j) {
    int lvl  = j + 4 * h + ((j >> 2) << 2);
    float s  = SCALE[lvl];
    float sx = px * s, sy = py * s, sz = pz * s;
    float fx = floorf(sx), fy = floorf(sy), fz = floorf(sz);
    float rx = sx - fx, ry = sy - fy, rz = sz - fz;
    unsigned ix0 = (unsigned)fx, iy0 = (unsigned)fy, iz0 = (unsigned)fz;
    unsigned ix1 = (unsigned)ceilf(sx), iy1 = (unsigned)ceilf(sy), iz1 = (unsigned)ceilf(sz);
    const float2* tl = (const float2*)table + (size_t)lvl * HT_SIZE;
    float f0 = 0.f, f1 = 0.f;
#pragma unroll
    for (int c = 0; c < 8; ++c) {
      unsigned cx = SELX[c] ? ix1 : ix0;
      unsigned cy = SELY[c] ? iy1 : iy0;
      unsigned cz = SELZ[c] ? iz1 : iz0;
      unsigned idx = (cx ^ cy * 2654435761u ^ cz * 805459861u) & (HT_SIZE - 1);
      float2 ft = tl[idx];
      float w = (SELX[c] ? rx : 1.f - rx) * (SELY[c] ? ry : 1.f - ry) *
                (SELZ[c] ? rz : 1.f - rz);
      f0 += w * ft.x;
      f1 += w * ft.y;
    }
    aenc[2 * j]     = (_Float16)f0;
    aenc[2 * j + 1] = (_Float16)f1;
  }

  // ---- density layer 1: relu(X @ Wd1 + b1) -> LDS (16x64 f16) -------------
  // C/D layout: lane holds n = m, rows M = i + 8h in vgpr i.
#pragma unroll
  for (int t = 0; t < 4; ++t) {
    float bb = b_d1[t * 16 + m];
    v8f c = {bb, bb, bb, bb, bb, bb, bb, bb};
    c = wmma_f16(aenc, BW[t * 32 + lane], c);
#pragma unroll
    for (int i = 0; i < 8; ++i)
      Hl[(i + 8 * h) * 64 + t * 16 + m] = (_Float16)fmaxf(c[i], 0.f);
  }

  // A fragments for 64-wide input: lane(m,h) reads row m, contiguous 8-half
  // chunks at k = {8h, 16+8h, 32+8h, 48+8h}.
  const v8h* hr = (const v8h*)(Hl + m * 64);
  v16h a0 = __builtin_shufflevector(hr[h], hr[2 + h],
                                    0,1,2,3,4,5,6,7,8,9,10,11,12,13,14,15);
  v16h a1 = __builtin_shufflevector(hr[4 + h], hr[6 + h],
                                    0,1,2,3,4,5,6,7,8,9,10,11,12,13,14,15);

  // ---- density layer 2 (no activation): write f32 out + stage f16 ---------
  {
    float bb = b_d2[m];
    v8f c = {bb, bb, bb, bb, bb, bb, bb, bb};
    c = wmma_f16(a0, BW[4 * 32 + lane], c);
    c = wmma_f16(a1, BW[5 * 32 + lane], c);
#pragma unroll
    for (int i = 0; i < 8; ++i) {
      int row = i + 8 * h;
      out_density[(size_t)(base + row) * 16 + m] = c[i];
      Hl[row * 16 + m] = (_Float16)c[i];
    }
  }

  // ---- color input fragment: [density cols | SH components] ---------------
  v8h d8 = *(const v8h*)(Hl + m * 16 + 8 * h);  // density cols 8h..8h+7 of row m
  float dx = dirp[gpt * 3 + 0], dy = dirp[gpt * 3 + 1], dz = dirp[gpt * 3 + 2];
  float xx = dx * dx, yy = dy * dy, zz = dz * dz;
  float sh[16];
  sh[0]  = 0.28209479177387814f;
  sh[1]  = 0.4886025119029199f * dy;
  sh[2]  = 0.4886025119029199f * dz;
  sh[3]  = 0.4886025119029199f * dx;
  sh[4]  = 1.0925484305920792f * dx * dy;
  sh[5]  = 1.0925484305920792f * dy * dz;
  sh[6]  = 0.9461746957575601f * zz - 0.31539156525252f;
  sh[7]  = 1.0925484305920792f * dx * dz;
  sh[8]  = 0.5462742152960396f * (xx - yy);
  sh[9]  = 0.5900435899266435f * dy * (3.f * xx - yy);
  sh[10] = 2.890611442640554f * dx * dy * dz;
  sh[11] = 0.4570457994644658f * dy * (5.f * zz - 1.f);
  sh[12] = 0.3731763325901154f * dz * (5.f * zz - 3.f);
  sh[13] = 0.4570457994644658f * dx * (5.f * zz - 1.f);
  sh[14] = 1.445305721320277f * dz * (xx - yy);
  sh[15] = 0.5900435899266435f * dx * (xx - 3.f * yy);
  v16h ac1;
#pragma unroll
  for (int e = 0; e < 8; ++e) {
    ac1[e]     = d8[e];                               // k = e + 8h  (density)
    ac1[8 + e] = (_Float16)(h ? sh[8 + e] : sh[e]);   // k = 16 + e + 8h (SH)
  }

  // ---- color layer 1: relu -> LDS (16x64) ---------------------------------
#pragma unroll
  for (int t = 0; t < 4; ++t) {
    float bb = b_c1[t * 16 + m];
    v8f c = {bb, bb, bb, bb, bb, bb, bb, bb};
    c = wmma_f16(ac1, BW[(6 + t) * 32 + lane], c);
#pragma unroll
    for (int i = 0; i < 8; ++i)
      Hl[(i + 8 * h) * 64 + t * 16 + m] = (_Float16)fmaxf(c[i], 0.f);
  }

  // ---- color layer 2: relu -> LDS (16x64) ---------------------------------
  v16h b0 = __builtin_shufflevector(hr[h], hr[2 + h],
                                    0,1,2,3,4,5,6,7,8,9,10,11,12,13,14,15);
  v16h b1 = __builtin_shufflevector(hr[4 + h], hr[6 + h],
                                    0,1,2,3,4,5,6,7,8,9,10,11,12,13,14,15);
#pragma unroll
  for (int t = 0; t < 4; ++t) {
    float bb = b_c2[t * 16 + m];
    v8f c = {bb, bb, bb, bb, bb, bb, bb, bb};
    c = wmma_f16(b0, BW[(10 + 2 * t) * 32 + lane], c);
    c = wmma_f16(b1, BW[(10 + 2 * t + 1) * 32 + lane], c);
#pragma unroll
    for (int i = 0; i < 8; ++i)
      Hl[(i + 8 * h) * 64 + t * 16 + m] = (_Float16)fmaxf(c[i], 0.f);
  }

  // ---- color layer 3 (cols 0-2 valid) + sigmoid ---------------------------
  v16h e0 = __builtin_shufflevector(hr[h], hr[2 + h],
                                    0,1,2,3,4,5,6,7,8,9,10,11,12,13,14,15);
  v16h e1 = __builtin_shufflevector(hr[4 + h], hr[6 + h],
                                    0,1,2,3,4,5,6,7,8,9,10,11,12,13,14,15);
  float bb3 = (m < 3) ? b_c3[m] : 0.f;
  v8f cc = {bb3, bb3, bb3, bb3, bb3, bb3, bb3, bb3};
  cc = wmma_f16(e0, BW[18 * 32 + lane], cc);
  cc = wmma_f16(e1, BW[19 * 32 + lane], cc);
  if (m < 3) {
#pragma unroll
    for (int i = 0; i < 8; ++i) {
      float sg = 1.f / (1.f + __expf(-cc[i]));
      out_color[(size_t)(base + i + 8 * h) * 3 + m] = sg;
    }
  }
}

extern "C" void kernel_launch(void* const* d_in, const int* in_sizes, int n_in,
                              void* d_out, int out_size, void* d_ws, size_t ws_size,
                              hipStream_t stream) {
  const float* pos   = (const float*)d_in[0];
  const float* dirp  = (const float*)d_in[1];
  const float* table = (const float*)d_in[2];
  const float* wd1   = (const float*)d_in[3];
  const float* bd1   = (const float*)d_in[4];
  const float* wd2   = (const float*)d_in[5];
  const float* bd2   = (const float*)d_in[6];
  const float* wc1   = (const float*)d_in[7];
  const float* bc1   = (const float*)d_in[8];
  const float* wc2   = (const float*)d_in[9];
  const float* bc2   = (const float*)d_in[10];
  const float* wc3   = (const float*)d_in[11];
  const float* bc3   = (const float*)d_in[12];
  int N = in_sizes[0] / 3;

  _Float16* wpack = (_Float16*)d_ws;         // 20 tiles * 512 halves = 20 KB
  float* out_density = (float*)d_out;        // N x 16
  float* out_color   = out_density + (size_t)N * 16;  // N x 3

  nerf_pack_weights<<<(20 * 512 + 255) / 256, 256, 0, stream>>>(
      wd1, wd2, wc1, wc2, wc3, wpack);

  int tiles  = (N + 15) / 16;                // 16 points per wave
  int blocks = (tiles + 7) / 8;              // 8 waves per block
  nerf_fused<<<blocks, 256, 0, stream>>>(pos, dirp, table, wpack, bd1, bd2,
                                         bc1, bc2, bc3, out_density, out_color, N);
}
